// myRWKV_86612310491227
// MI455X (gfx1250) — matrix operand order repaired and checked by reading, not compiled, not run on hardware
//
#include <hip/hip_runtime.h>

// ---------------- problem constants ----------------
constexpr int L  = 12;
constexpr int D  = 768;
constexpr int FF = 3072;
constexpr int V  = 50277;
constexpr int T  = 2048;
constexpr float EPS = 1e-5f;

// ---------------- vector types ----------------
typedef __attribute__((ext_vector_type(16))) _Float16 v16h;
typedef __attribute__((ext_vector_type(8)))  _Float16 v8h;
typedef __attribute__((ext_vector_type(4)))  _Float16 v4h;
typedef __attribute__((ext_vector_type(8)))  float    v8f;
typedef __attribute__((ext_vector_type(4)))  float    v4f;

union H16 { v16h v; v8h h[2]; };

// ================= block-wide reduction (256 threads) =================
__device__ __forceinline__ float block_sum(float val, float* sm) {
    int t = threadIdx.x;
    sm[t] = val;
    __syncthreads();
    #pragma unroll
    for (int s = 128; s > 0; s >>= 1) {
        if (t < s) sm[t] += sm[t + s];
        __syncthreads();
    }
    float r = sm[0];
    __syncthreads();
    return r;
}

// ================= embedding + LN =================
__global__ __launch_bounds__(256) void embed_ln_kernel(
    const int* __restrict__ token, const float* __restrict__ emb,
    const float* __restrict__ w, const float* __restrict__ b,
    float* __restrict__ x)
{
    __shared__ float sm[256];
    int t = blockIdx.x;
    const float* row = emb + (size_t)token[t] * D;
    float e[3];
    #pragma unroll
    for (int j = 0; j < 3; ++j) e[j] = row[threadIdx.x + j * 256];
    float m = block_sum(e[0] + e[1] + e[2], sm) * (1.0f / D);
    float vs = 0.f;
    #pragma unroll
    for (int j = 0; j < 3; ++j) { float d0 = e[j] - m; vs += d0 * d0; }
    float var = block_sum(vs, sm) * (1.0f / D);
    float rs = rsqrtf(var + EPS);
    #pragma unroll
    for (int j = 0; j < 3; ++j) {
        int d0 = threadIdx.x + j * 256;
        x[(size_t)t * D + d0] = (e[j] - m) * rs * w[d0] + b[d0];
    }
}

// ================= generic LayerNorm over rows of D =================
__global__ __launch_bounds__(256) void ln_kernel(
    const float* __restrict__ in, const float* __restrict__ w,
    const float* __restrict__ b, float* __restrict__ out)
{
    __shared__ float sm[256];
    const float* row = in + (size_t)blockIdx.x * D;
    float e[3];
    #pragma unroll
    for (int j = 0; j < 3; ++j) e[j] = row[threadIdx.x + j * 256];
    float m = block_sum(e[0] + e[1] + e[2], sm) * (1.0f / D);
    float vs = 0.f;
    #pragma unroll
    for (int j = 0; j < 3; ++j) { float d0 = e[j] - m; vs += d0 * d0; }
    float var = block_sum(vs, sm) * (1.0f / D);
    float rs = rsqrtf(var + EPS);
    #pragma unroll
    for (int j = 0; j < 3; ++j) {
        int d0 = threadIdx.x + j * 256;
        out[(size_t)blockIdx.x * D + d0] = (e[j] - m) * rs * w[d0] + b[d0];
    }
}

// ================= time-shift lerp (attention, 3 outputs, f16) =================
__global__ __launch_bounds__(256) void lerp3_att_kernel(
    const float* __restrict__ xy, const float* __restrict__ statea,
    const float* __restrict__ wk, const float* __restrict__ wv,
    const float* __restrict__ wr,
    _Float16* __restrict__ kin, _Float16* __restrict__ vin,
    _Float16* __restrict__ rin, float* __restrict__ ot_xy)
{
    int t = blockIdx.x;
    #pragma unroll
    for (int j = 0; j < 3; ++j) {
        int d0 = threadIdx.x + j * 256;
        size_t idx = (size_t)t * D + d0;
        float cur = xy[idx];
        float prv = (t == 0) ? statea[d0] : xy[idx - D];
        kin[idx] = (_Float16)(prv + wk[d0] * (cur - prv));
        vin[idx] = (_Float16)(prv + wv[d0] * (cur - prv));
        rin[idx] = (_Float16)(prv + wr[d0] * (cur - prv));
        if (t == T - 1) ot_xy[d0] = cur;
    }
}

// ================= time-shift lerp (FFN, 2 outputs, f16) =================
__global__ __launch_bounds__(256) void lerp2_ffn_kernel(
    const float* __restrict__ ddd, const float* __restrict__ stated,
    const float* __restrict__ wk, const float* __restrict__ wr,
    _Float16* __restrict__ kin, _Float16* __restrict__ rin,
    float* __restrict__ ot_ddd)
{
    int t = blockIdx.x;
    #pragma unroll
    for (int j = 0; j < 3; ++j) {
        int d0 = threadIdx.x + j * 256;
        size_t idx = (size_t)t * D + d0;
        float cur = ddd[idx];
        float prv = (t == 0) ? stated[d0] : ddd[idx - D];
        kin[idx] = (_Float16)(prv + wk[d0] * (cur - prv));
        rin[idx] = (_Float16)(prv + wr[d0] * (cur - prv));
        if (t == T - 1) ot_ddd[d0] = cur;
    }
}

// ================= WKV serial scan (one lane per channel) =================
__global__ __launch_bounds__(128) void wkv_scan_kernel(
    const float* __restrict__ k, const float* __restrict__ v,
    const float* __restrict__ tf, const float* __restrict__ td,
    const float* __restrict__ sb0, const float* __restrict__ sc0,
    float* __restrict__ rz, float* __restrict__ ot_sb, float* __restrict__ ot_sc)
{
    int d0 = blockIdx.x * blockDim.x + threadIdx.x;
    if (d0 >= D) return;
    float sb = sb0[d0], sc = sc0[d0];
    float f = tf[d0], w = td[d0];
    for (int t = 0; t < T; ++t) {
        size_t idx = (size_t)t * D + d0;
        float kt = k[idx], vt = v[idx];
        float kv = kt * vt;
        rz[idx] = (sb + kv * f) / (sc + kt * f);
        sb = w * sb + w * kv;   // single-FMA critical path per step
        sc = w * sc + w * kt;
    }
    ot_sb[d0] = sb;
    ot_sc[d0] = sc;
}

// ================= elementwise helpers (vectorized x4) =================
__global__ void conv_f16_kernel(const float* __restrict__ in,
                                _Float16* __restrict__ out, int n4)
{
    const v4f* in4 = (const v4f*)in;
    v4h* out4 = (v4h*)out;
    for (int i = blockIdx.x * blockDim.x + threadIdx.x; i < n4;
         i += gridDim.x * blockDim.x) {
        v4f a = in4[i];
        v4h o;
        #pragma unroll
        for (int j = 0; j < 4; ++j) o[j] = (_Float16)a[j];
        out4[i] = o;
    }
}

__global__ void premul_f16_kernel(const float* __restrict__ a,
                                  const float* __restrict__ b,
                                  _Float16* __restrict__ out, int n4)
{
    const v4f* a4 = (const v4f*)a;
    const v4f* b4 = (const v4f*)b;
    v4h* out4 = (v4h*)out;
    for (int i = blockIdx.x * blockDim.x + threadIdx.x; i < n4;
         i += gridDim.x * blockDim.x) {
        v4f x = a4[i], y = b4[i];
        v4h o;
        #pragma unroll
        for (int j = 0; j < 4; ++j) o[j] = (_Float16)(x[j] * y[j]);
        out4[i] = o;
    }
}

// ================= WMMA GEMM: C[M,N] = A[M,K] @ W[N,K]^T =================
// Block = 256 threads = 8 waves, block tile 128(M) x 128(N).
// Wave (wm,wn): 64(M) x 32(N) -> 4x2 WMMA tiles, 8 v_wmma per K-step of 32.
// A tile (128x32 f16) staged per K-step into LDS via global_load_async_to_lds
// (ASYNCcnt), double-buffered; fragments come from ds_load_b128.
// B fragments double-buffered in registers (disjoint sets b0/b1, two-phase
// unrolled K loop -> no rolling copies). Raw split barrier (no loadcnt fence)
// so in-flight B loads cross the barrier.

#define LDSA_STRIDE 40   // halves per row (32 data + 8 pad)

// LDS producer/consumer sync: wait own async writes, then wg split barrier.
// Deliberately NOT __syncthreads(): no loadcnt fence, so register-destined
// global loads stay in flight across the barrier.
#define GEMM_BARRIER() \
    asm volatile("s_wait_asynccnt 0x0\n\t" \
                 "s_barrier_signal -1\n\t" \
                 "s_barrier_wait -1" ::: "memory")

// Epilogue modes
#define EP_NONE    0
#define EP_EXP     1
#define EP_SIGMOID 2
#define EP_RELU2H  3
#define EP_ADD     4
#define EP_MULADD  5

template <int EP>
__global__ __launch_bounds__(256) void gemm_wmma_kernel(
    const _Float16* __restrict__ A, const _Float16* __restrict__ W,
    float* __restrict__ C, _Float16* __restrict__ Ch,
    const float* __restrict__ base, const float* __restrict__ mul,
    int M, int N, int K)
{
    __shared__ alignas(16) _Float16 sA[2][128 * LDSA_STRIDE];

    const int tid  = threadIdx.x;
    const int wave = tid >> 5;
    const int lane = tid & 31;
    const int l15  = lane & 15;
    const int hi   = lane >> 4;
    const int wm   = wave >> 2;      // 0..1
    const int wn   = wave & 3;       // 0..3
    const int mBase = blockIdx.y * 128;
    const int nBase = blockIdx.x * 128;
    const int mWave = mBase + wm * 64;
    const int nWave = nBase + wn * 32;

    // ---- async stage of A tile [128 x 32] for K-chunk kk into buffer p ----
    auto stage_async = [&](int p, int kk) {
        #pragma unroll
        for (int i = 0; i < 2; ++i) {
            int e   = tid + i * 256;          // 0..511 -> 512 x 16B = 8KB
            int row = e >> 2;
            int c   = e & 3;
            const _Float16* g = A + (size_t)(mBase + row) * K + kk + c * 8;
            unsigned lds = (unsigned)(unsigned long long)
                           (const void*)&sA[p][row * LDSA_STRIDE + c * 8];
            asm volatile("global_load_async_to_lds_b128 %0, %1, off"
                         :: "v"(lds), "v"(g) : "memory");
        }
    };

    // ---- B fragments (2 N-tiles) from global ----
    auto loadB = [&](int kk, v16h* bf) {
        #pragma unroll
        for (int jn = 0; jn < 2; ++jn) {
            const _Float16* p = W + (size_t)(nWave + jn * 16 + l15) * K + kk + hi * 8;
            H16 u;
            u.h[0] = *(const v8h*)(p);
            u.h[1] = *(const v8h*)(p + 16);
            bf[jn] = u.v;
        }
    };

    // ---- A fragments (4 M-tiles) from LDS buffer p ----
    auto loadA = [&](int p, v16h* af) {
        #pragma unroll
        for (int i = 0; i < 4; ++i) {
            int row = wm * 64 + i * 16 + l15;
            const _Float16* q = &sA[p][row * LDSA_STRIDE + hi * 8];
            H16 u;
            u.h[0] = *(const v8h*)(q);
            u.h[1] = *(const v8h*)(q + 16);
            af[i] = u.v;
        }
    };

    v8f acc[4][2];
    #pragma unroll
    for (int i = 0; i < 4; ++i)
        #pragma unroll
        for (int jn = 0; jn < 2; ++jn) acc[i][jn] = {};

    v16h b0[2] = {}, b1[2] = {}, af[4];

    stage_async(0, 0);
    loadB(0, b0);
    GEMM_BARRIER();

    for (int kk = 0; kk < K; kk += 64) {
        // ---- phase 0: compute on buf0/b0, stage buf1/b1 for kk+32 ----
        stage_async(1, kk + 32);       // kk+32 < K always (K % 64 == 0)
        loadB(kk + 32, b1);
        if (kk + 128 < K) {
            #pragma unroll
            for (int jn = 0; jn < 2; ++jn)
                __builtin_prefetch(W + (size_t)(nWave + jn * 16 + l15) * K + kk + 128, 0, 1);
        }
        loadA(0, af);
        #pragma unroll
        for (int i = 0; i < 4; ++i)
            #pragma unroll
            for (int jn = 0; jn < 2; ++jn)
                acc[i][jn] = __builtin_amdgcn_wmma_f32_16x16x32_f16(
                    false, af[i], false, b0[jn], (short)0, acc[i][jn], false, false);
        GEMM_BARRIER();

        // ---- phase 1: compute on buf1/b1, stage buf0/b0 for kk+64 ----
        if (kk + 64 < K) {
            stage_async(0, kk + 64);
            loadB(kk + 64, b0);
            if (kk + 160 < K) {
                #pragma unroll
                for (int jn = 0; jn < 2; ++jn)
                    __builtin_prefetch(W + (size_t)(nWave + jn * 16 + l15) * K + kk + 160, 0, 1);
            }
        }
        loadA(1, af);
        #pragma unroll
        for (int i = 0; i < 4; ++i)
            #pragma unroll
            for (int jn = 0; jn < 2; ++jn)
                acc[i][jn] = __builtin_amdgcn_wmma_f32_16x16x32_f16(
                    false, af[i], false, b1[jn], (short)0, acc[i][jn], false, false);
        GEMM_BARRIER();
    }

    // ---- epilogue ----
    #pragma unroll
    for (int i = 0; i < 4; ++i) {
        #pragma unroll
        for (int jn = 0; jn < 2; ++jn) {
            int col = nWave + jn * 16 + l15;
            #pragma unroll
            for (int j = 0; j < 8; ++j) {
                int row = mWave + i * 16 + hi * 8 + j;
                size_t idx = (size_t)row * N + col;
                float v0 = acc[i][jn][j];
                if (EP == EP_NONE) {
                    C[idx] = v0;
                } else if (EP == EP_EXP) {
                    C[idx] = __expf(v0);
                } else if (EP == EP_SIGMOID) {
                    C[idx] = 1.0f / (1.0f + __expf(-v0));
                } else if (EP == EP_RELU2H) {
                    float r = v0 > 0.f ? v0 : 0.f;
                    Ch[idx] = (_Float16)(r * r);
                } else if (EP == EP_ADD) {
                    C[idx] = base[idx] + v0;
                } else if (EP == EP_MULADD) {
                    C[idx] = base[idx] + v0 * mul[idx];
                }
            }
        }
    }
}

// ================= head GEMV for the last token =================
__global__ __launch_bounds__(256) void head_gemv_kernel(
    const float* __restrict__ xf, const float* __restrict__ head,
    float* __restrict__ logits)
{
    __shared__ float sxf[D];
    #pragma unroll
    for (int j = 0; j < 3; ++j) sxf[threadIdx.x + j * 256] = xf[threadIdx.x + j * 256];
    __syncthreads();

    int wave = threadIdx.x >> 5;
    int lane = threadIdx.x & 31;
    #pragma unroll
    for (int rr = 0; rr < 4; ++rr) {
        int row = blockIdx.x * 32 + wave * 4 + rr;
        if (row < V) {
            const float* hrow = head + (size_t)row * D;
            float s = 0.f;
            #pragma unroll
            for (int j = 0; j < D / 32; ++j)
                s += hrow[lane + j * 32] * sxf[lane + j * 32];
            #pragma unroll
            for (int o = 16; o > 0; o >>= 1)
                s += __shfl_xor(s, o, 32);
            if (lane == 0) logits[row] = s;
        }
    }
}

// =====================================================================
extern "C" void kernel_launch(void* const* d_in, const int* in_sizes, int n_in,
                              void* d_out, int out_size, void* d_ws, size_t ws_size,
                              hipStream_t stream)
{
    (void)in_sizes; (void)n_in; (void)out_size; (void)ws_size;

    const int*   token      = (const int*)  d_in[0];
    const float* state      = (const float*)d_in[1];
    const float* emb        = (const float*)d_in[2];
    const float* emb_ln_w   = (const float*)d_in[3];
    const float* emb_ln_b   = (const float*)d_in[4];
    const float* ln1w       = (const float*)d_in[5];
    const float* ln1b       = (const float*)d_in[6];
    const float* ln2w       = (const float*)d_in[7];
    const float* ln2b       = (const float*)d_in[8];
    const float* time_decay = (const float*)d_in[9];
    const float* time_first = (const float*)d_in[10];
    const float* kktk       = (const float*)d_in[11];
    const float* vvtv       = (const float*)d_in[12];
    const float* rrtr       = (const float*)d_in[13];
    const float* att_key    = (const float*)d_in[14];
    const float* att_value  = (const float*)d_in[15];
    const float* att_recept = (const float*)d_in[16];
    const float* att_out    = (const float*)d_in[17];
    const float* tmk_ffn    = (const float*)d_in[18];
    const float* tmr_ffn    = (const float*)d_in[19];
    const float* key_ffn    = (const float*)d_in[20];
    const float* recept_ffn = (const float*)d_in[21];
    const float* value_ffn  = (const float*)d_in[22];
    const float* ln_out_w   = (const float*)d_in[23];
    const float* ln_out_b   = (const float*)d_in[24];
    const float* head       = (const float*)d_in[25];

    float* out        = (float*)d_out;
    float* out_logits = out;          // V floats
    float* out_ot     = out + V;      // 48 x D floats

    // ---- workspace carve-out (256B aligned) ----
    char* ws = (char*)d_ws;
    size_t off = 0;
    auto take = [&](size_t bytes) -> char* {
        char* p = ws + off;
        off += (bytes + 255) & ~(size_t)255;
        return p;
    };
    _Float16* wk_h    = (_Float16*)take((size_t)D * D * 2);
    _Float16* wv_h    = (_Float16*)take((size_t)D * D * 2);
    _Float16* wr_h    = (_Float16*)take((size_t)D * D * 2);
    _Float16* wo_h    = (_Float16*)take((size_t)D * D * 2);
    _Float16* wkffn_h = (_Float16*)take((size_t)FF * D * 2);
    _Float16* wrffn_h = (_Float16*)take((size_t)D * D * 2);
    _Float16* wvffn_h = (_Float16*)take((size_t)D * FF * 2);

    float*    x    = (float*)   take((size_t)T * D * 4);
    float*    xy   = (float*)   take((size_t)T * D * 4);
    _Float16* kin  = (_Float16*)take((size_t)T * D * 2);
    _Float16* vin  = (_Float16*)take((size_t)T * D * 2);
    _Float16* rin  = (_Float16*)take((size_t)T * D * 2);
    float*    kbuf = (float*)   take((size_t)T * D * 4);
    float*    vbuf = (float*)   take((size_t)T * D * 4);
    float*    rbuf = (float*)   take((size_t)T * D * 4);
    float*    rz   = (float*)   take((size_t)T * D * 4);
    _Float16* rrz  = (_Float16*)take((size_t)T * D * 2);
    float*    mvv  = (float*)   take((size_t)T * D * 4);
    float*    ddd  = (float*)   take((size_t)T * D * 4);
    _Float16* kmin = (_Float16*)take((size_t)T * D * 2);
    _Float16* rtin = (_Float16*)take((size_t)T * D * 2);
    _Float16* hbuf = (_Float16*)take((size_t)T * FF * 2);
    float*    rt   = (float*)   take((size_t)T * D * 4);
    float*    xf   = (float*)   take((size_t)D * 4);

    const dim3 blk256(256);
    const dim3 gemmD(D / 128, T / 128);    // N=768:  (6,16)
    const dim3 gemmFF(FF / 128, T / 128);  // N=3072: (24,16)

    // ---- embedding + LN ----
    embed_ln_kernel<<<T, blk256, 0, stream>>>(token, emb, emb_ln_w, emb_ln_b, x);

    for (int i = 0; i < L; ++i) {
        const float* statea = state + (size_t)(4 * i + 0) * D;
        const float* stateb = state + (size_t)(4 * i + 1) * D;
        const float* statec = state + (size_t)(4 * i + 2) * D;
        const float* stated = state + (size_t)(4 * i + 3) * D;

        // convert this layer's weights to f16 (vectorized x4)
        conv_f16_kernel<<<1024, 256, 0, stream>>>(att_key    + (size_t)i * D * D, wk_h,    D * D / 4);
        conv_f16_kernel<<<1024, 256, 0, stream>>>(att_value  + (size_t)i * D * D, wv_h,    D * D / 4);
        conv_f16_kernel<<<1024, 256, 0, stream>>>(att_recept + (size_t)i * D * D, wr_h,    D * D / 4);
        conv_f16_kernel<<<1024, 256, 0, stream>>>(att_out    + (size_t)i * D * D, wo_h,    D * D / 4);
        conv_f16_kernel<<<1024, 256, 0, stream>>>(key_ffn    + (size_t)i * FF * D, wkffn_h, FF * D / 4);
        conv_f16_kernel<<<1024, 256, 0, stream>>>(recept_ffn + (size_t)i * D * D, wrffn_h, D * D / 4);
        conv_f16_kernel<<<1024, 256, 0, stream>>>(value_ffn  + (size_t)i * D * FF, wvffn_h, D * FF / 4);

        // LN1 + time-shift lerp
        ln_kernel<<<T, blk256, 0, stream>>>(x, ln1w + (size_t)i * D, ln1b + (size_t)i * D, xy);
        lerp3_att_kernel<<<T, blk256, 0, stream>>>(
            xy, statea, kktk + (size_t)i * D, vvtv + (size_t)i * D, rrtr + (size_t)i * D,
            kin, vin, rin, out_ot + (size_t)(4 * i + 0) * D);

        // attention projections (WMMA): k = exp(.), v = ., r = sigmoid(.)
        gemm_wmma_kernel<EP_EXP><<<gemmD, blk256, 0, stream>>>(
            kin, wk_h, kbuf, nullptr, nullptr, nullptr, T, D, D);
        gemm_wmma_kernel<EP_NONE><<<gemmD, blk256, 0, stream>>>(
            vin, wv_h, vbuf, nullptr, nullptr, nullptr, T, D, D);
        gemm_wmma_kernel<EP_SIGMOID><<<gemmD, blk256, 0, stream>>>(
            rin, wr_h, rbuf, nullptr, nullptr, nullptr, T, D, D);

        // serial WKV scan (also emits final sb, sc state rows)
        wkv_scan_kernel<<<D / 128, 128, 0, stream>>>(
            kbuf, vbuf, time_first + (size_t)i * D, time_decay + (size_t)i * D,
            stateb, statec, rz,
            out_ot + (size_t)(4 * i + 1) * D, out_ot + (size_t)(4 * i + 2) * D);

        // mvv = x + (r * rz) @ att_out^T
        premul_f16_kernel<<<1024, 256, 0, stream>>>(rbuf, rz, rrz, T * D / 4);
        gemm_wmma_kernel<EP_ADD><<<gemmD, blk256, 0, stream>>>(
            rrz, wo_h, mvv, nullptr, x, nullptr, T, D, D);

        // channel mix
        ln_kernel<<<T, blk256, 0, stream>>>(mvv, ln2w + (size_t)i * D, ln2b + (size_t)i * D, ddd);
        lerp2_ffn_kernel<<<T, blk256, 0, stream>>>(
            ddd, stated, tmk_ffn + (size_t)i * D, tmr_ffn + (size_t)i * D,
            kmin, rtin, out_ot + (size_t)(4 * i + 3) * D);

        // h = relu(km @ key_ffn^T)^2  (f16 out), rt = sigmoid(. @ recept_ffn^T)
        gemm_wmma_kernel<EP_RELU2H><<<gemmFF, blk256, 0, stream>>>(
            kmin, wkffn_h, nullptr, hbuf, nullptr, nullptr, T, FF, D);
        gemm_wmma_kernel<EP_SIGMOID><<<gemmD, blk256, 0, stream>>>(
            rtin, wrffn_h, rt, nullptr, nullptr, nullptr, T, D, D);

        // x = mvv + (h @ value_ffn^T) * rt
        gemm_wmma_kernel<EP_MULADD><<<gemmD, blk256, 0, stream>>>(
            hbuf, wvffn_h, x, nullptr, mvv, rt, T, D, FF);
    }

    // final LN of last token + head GEMV
    ln_kernel<<<1, blk256, 0, stream>>>(x + (size_t)(T - 1) * D, ln_out_w, ln_out_b, xf);
    head_gemv_kernel<<<(V + 31) / 32, blk256, 0, stream>>>(xf, head, out_logits);
}
